// HANCL_17437567221851
// MI455X (gfx1250) — matrix operand order repaired and checked by reading, not compiled, not run on hardware
//
#include <hip/hip_runtime.h>
#include <hip/hip_bf16.h>
#include <math.h>

// ---------------------------------------------------------------------------
// HANCL forward for MI455X (gfx1250, wave32, WMMA).
// DIM=64, N_LAYER=2, B=2048, M=64, TEMP=0.2, KGE_W=1e-6
// ---------------------------------------------------------------------------

#define B_SZ   2048
#define M_SZ   64
#define D_SZ   64
#define LSTRIDE (B_SZ * M_SZ)      // idx layer stride
#define ROWS   (3 * B_SZ)          // 6144 rows per stack
#define INV_T  5.0f                // 1/TEMP

// LDS row strides (in halves): multiples of 8 -> 16B-aligned lane runs
// (ds_load_b128-able) and conflict-free quad-bank spacing on 64 banks.
#define XSTR   136                 // 128-wide rows (x, tang, w1^T)
#define NSTR   72                  // 64-wide rows (NA/NB tiles)

typedef _Float16 v16h __attribute__((ext_vector_type(16)));
typedef float    v8f  __attribute__((ext_vector_type(8)));

__device__ __forceinline__ float dev_artanh(float x) {
    x = fminf(fmaxf(x, -1.0f + 1e-5f), 1.0f - 1e-5f);
    return atanhf(x);
}

// ---- WMMA fragment loaders (layouts per CDNA5 ISA 7.12.2) ------
// A: 16x32 f16, row-major source with given stride (halves).
// Per lane: two contiguous 8-half (16 B) runs -> 2x ds_load_b128.
__device__ __forceinline__ v16h load_a_frag(const _Float16* lds, int row0, int k0, int stride) {
    int lane = threadIdx.x & 31;
    const _Float16* p = lds + (row0 + (lane & 15)) * stride;
    int kb = k0 + ((lane & 16) ? 8 : 0);
    v16h a;
#pragma unroll
    for (int j = 0; j < 16; ++j) {
        int v = j >> 1;
        int k = kb + ((v & 4) ? 16 : 0) + ((v & 3) << 1) + (j & 1);
        a[j] = p[k];
    }
    return a;
}
// B: 32x16 f16 where B[k][n] = src[n][k] (src row-major [N][K]) -> A @ src^T.
// Per lane: two contiguous 8-half (16 B) runs -> 2x ds_load_b128.
__device__ __forceinline__ v16h load_b_frag_nk(const _Float16* lds, int n0, int k0, int stride) {
    int lane = threadIdx.x & 31;
    const _Float16* p = lds + (n0 + (lane & 15)) * stride + k0 + ((lane & 16) ? 8 : 0);
    v16h b;
#pragma unroll
    for (int j = 0; j < 16; ++j) b[j] = p[(j < 8) ? j : j + 8];
    return b;
}

// ---------------------------------------------------------------------------
// init embeddings
// ---------------------------------------------------------------------------
__global__ void k_mean_init(const float* __restrict__ E, const int* __restrict__ idx,
                            float* __restrict__ out) {
    int b = blockIdx.x, d = threadIdx.x;               // 64 threads
    const int* p = idx + (size_t)b * M_SZ;             // layer-0 slice
    float acc = 0.f;
    for (int m = 0; m < M_SZ; ++m) acc += E[(size_t)p[m] * D_SZ + d];
    out[(size_t)b * D_SZ + d] = acc * (1.0f / 64.0f);
}

__global__ void k_item_init(const float* __restrict__ E, const int* __restrict__ items,
                            float* __restrict__ out) {
    int b = blockIdx.x, d = threadIdx.x;
    out[(size_t)b * D_SZ + d] = E[(size_t)items[b] * D_SZ + d];
}

// ---------------------------------------------------------------------------
// Euclidean attention aggregation (one block per batch element, 128 thr)
// ---------------------------------------------------------------------------
__global__ void k_ed_agg(const float* __restrict__ E, const float* __restrict__ R,
                         const int* __restrict__ hidx, const int* __restrict__ ridx,
                         const int* __restrict__ tidx, const float* __restrict__ w1,
                         const float* __restrict__ w2, float* __restrict__ out, int layer) {
    __shared__ __align__(16) _Float16 sx[M_SZ * XSTR];   // x = [h | path], f16
    __shared__ __align__(16) _Float16 sw1t[64 * XSTR];   // att_w1 transposed: [n][k]
    __shared__ float sw2[64];
    __shared__ float satt[64];
    __shared__ float swgt[64];

    int b = blockIdx.x, t = threadIdx.x;
    const int* h0 = hidx + (size_t)b * M_SZ;
    const int* h1 = hidx + LSTRIDE + (size_t)b * M_SZ;
    const int* r0 = ridx + (size_t)b * M_SZ;
    const int* r1 = ridx + LSTRIDE + (size_t)b * M_SZ;

    for (int i = t; i < 64 * 128; i += 128) {            // w1[k][n] -> sw1t[n][k]
        int n = i >> 7, k = i & 127;
        sw1t[n * XSTR + k] = (_Float16)w1[k * 64 + n];
    }
    if (t < 64) sw2[t] = w2[t];

    for (int i = t; i < M_SZ * 128; i += 128) {
        int m = i >> 7, k = i & 127;
        float v;
        if (k < 64) {
            v = E[(size_t)h0[m] * D_SZ + k];
            if (layer == 1) v += E[(size_t)h1[m] * D_SZ + k];
        } else {
            int kk = k - 64;
            v = R[(size_t)r0[m] * D_SZ + kk];
            if (layer == 1) v *= R[(size_t)r1[m] * D_SZ + kk];
        }
        sx[m * XSTR + k] = (_Float16)v;
    }
    __syncthreads();

    // a = sigmoid(relu(x @ w1) @ w2), via WMMA; wave wv owns rows 16*wv..
    int wv = t >> 5, lane = t & 31, row0 = wv * 16;
    float dot[8];
#pragma unroll
    for (int r = 0; r < 8; ++r) dot[r] = 0.f;
#pragma unroll
    for (int nt = 0; nt < 4; ++nt) {
        v8f acc = {};
#pragma unroll
        for (int kc = 0; kc < 4; ++kc) {
            v16h a  = load_a_frag(sx, row0, kc * 32, XSTR);
            v16h bb = load_b_frag_nk(sw1t, nt * 16, kc * 32, XSTR);
            acc = __builtin_amdgcn_wmma_f32_16x16x32_f16(false, a, false, bb,
                                                         (short)0, acc, false, false);
        }
        float wc = sw2[nt * 16 + (lane & 15)];
#pragma unroll
        for (int r = 0; r < 8; ++r) dot[r] += fmaxf(acc[r], 0.f) * wc;
    }
#pragma unroll
    for (int r = 0; r < 8; ++r) {
        float s = dot[r];
        s += __shfl_xor(s, 1, 32); s += __shfl_xor(s, 2, 32);
        s += __shfl_xor(s, 4, 32); s += __shfl_xor(s, 8, 32);
        if ((lane & 15) == 0)
            satt[row0 + r + ((lane & 16) ? 8 : 0)] = 1.f / (1.f + expf(-s));
    }
    __syncthreads();

    if (t < 64) {
        float mx = -1e30f;
        for (int m = 0; m < M_SZ; ++m) mx = fmaxf(mx, satt[m]);
        float sum = 0.f;
        for (int m = 0; m < M_SZ; ++m) sum += expf(satt[m] - mx);
        swgt[t] = expf(satt[t] - mx) / sum;
    }
    __syncthreads();

    if (t < 64) {
        const int* ti = tidx + (size_t)layer * LSTRIDE + (size_t)b * M_SZ;
        float acc = 0.f;
        for (int m = 0; m < M_SZ; ++m) acc += swgt[m] * E[(size_t)ti[m] * D_SZ + t];
        out[(size_t)b * D_SZ + t] = acc;
    }
}

// ---------------------------------------------------------------------------
// Hyperbolic attention aggregation (one block per batch element, 128 thr)
// All norms/dots of the exp/log maps are analytic in registers; only the
// linear-combination vectors (hh, ht, tang, e) touch LDS.
// ---------------------------------------------------------------------------
__global__ void k_hyper_agg(const float* __restrict__ E, const float* __restrict__ R,
                            const int* __restrict__ hidx, const int* __restrict__ ridx,
                            const int* __restrict__ tidx,
                            const float* __restrict__ w1, const float* __restrict__ b1,
                            const float* __restrict__ w2, const float* __restrict__ b2p,
                            const float* __restrict__ curv,
                            float* __restrict__ out, int layer) {
    __shared__ float sHH[M_SZ * 65];                      // hh rows
    __shared__ float sHT[M_SZ * 65];                      // ht rows
    __shared__ __align__(16) char su[M_SZ * XSTR * 2];    // tang(f16) then e-contrib(f32 [64][65])
    __shared__ __align__(16) _Float16 sw1t[64 * XSTR];    // a2_w1 transposed: [n][k]
    __shared__ float sw2[64], sb1[64];
    __shared__ float satt[64], swgt[64], smean[64];

    _Float16* stang = (_Float16*)su;
    float*    sS    = (float*)su;

    int b = blockIdx.x, t = threadIdx.x;
    float c = fabsf(curv[0]);

    for (int i = t; i < 64 * 128; i += 128) {
        int n = i >> 7, k = i & 127;
        sw1t[n * XSTR + k] = (_Float16)w1[k * 64 + n];
    }
    if (t < 64) { sw2[t] = w2[t]; sb1[t] = b1[t]; }

    float ht2_reg = 0.f;  // ||ht_m||^2, kept across phases by thread m
    if (t < 64) {
        int m = t;
        const int* h0 = hidx + (size_t)b * M_SZ;
        const int* h1 = hidx + LSTRIDE + (size_t)b * M_SZ;
        const int* r0 = ridx + (size_t)b * M_SZ;
        const int* r1 = ridx + LSTRIDE + (size_t)b * M_SZ;
        const int* ti = tidx + (size_t)layer * LSTRIDE + (size_t)b * M_SZ;
        size_t eh0 = (size_t)h0[m] * D_SZ, eh1 = (size_t)h1[m] * D_SZ;
        size_t er0 = (size_t)r0[m] * D_SZ, er1 = (size_t)r1[m] * D_SZ;
        size_t et  = (size_t)ti[m] * D_SZ;

        // pass 1: gather + dot accumulation (rows stay in L2/L0 for pass 2)
        float h2 = 0, p2 = 0, t2 = 0, htd = 0, hpd = 0;
        for (int d = 0; d < D_SZ; ++d) {
            float hv = E[eh0 + d]; if (layer == 1) hv += E[eh1 + d];
            float pv = R[er0 + d]; if (layer == 1) pv *= R[er1 + d];
            float tv = E[et + d];
            h2 += hv * hv; p2 += pv * pv; t2 += tv * tv;
            htd += hv * tv; hpd += hv * pv;
        }
        float invh = 1.f / fmaxf(sqrtf(h2), 1e-12f);
        float invp = 1.f / fmaxf(sqrtf(p2), 1e-12f);
        float invt = 1.f / fmaxf(sqrtf(t2), 1e-12f);
        // hh = expmap0(l2(h)) * |c|
        float nh = sqrtf(fmaxf(h2 * invh * invh, 1e-15f));
        float scale_hh = tanhf(nh) / nh * invh * c;
        float hh2 = scale_hh * scale_hh * h2;
        float lam = 2.f / fmaxf(1.f - hh2, 1e-15f);
        // ht = expmap(l2(t), hh) = cA*h_raw + cB*t_raw
        float ntn = sqrtf(fmaxf(t2 * invt * invt, 1e-15f));
        float coef_t = tanhf(lam * ntn * 0.5f) / ntn * invt;
        float y2 = coef_t * coef_t * t2;
        float xy = scale_hh * coef_t * htd;
        float den = fmaxf(1.f + 2.f * xy + hh2 * y2, 1e-15f);
        float cA = (1.f + 2.f * xy + y2) * scale_hh / den;
        float cB = (1.f - hh2) * coef_t / den;
        ht2_reg = cA * cA * h2 + 2.f * cA * cB * htd + cB * cB * t2;
        // hr = expmap(l2(path), hh) = cAr*h_raw + cBr*p_raw
        float nrn = sqrtf(fmaxf(p2 * invp * invp, 1e-15f));
        float coef_r = tanhf(lam * nrn * 0.5f) / nrn * invp;
        float y2r = coef_r * coef_r * p2;
        float xyr = scale_hh * coef_r * hpd;
        float denr = fmaxf(1.f + 2.f * xyr + hh2 * y2r, 1e-15f);
        float cAr = (1.f + 2.f * xyr + y2r) * scale_hh / denr;
        float cBr = (1.f - hh2) * coef_r / denr;
        float hr2 = cAr * cAr * h2 + 2.f * cAr * cBr * hpd + cBr * cBr * p2;
        // tang = logmap0(concat(hh, hr)), norm over 128 dims
        float ncat = sqrtf(fmaxf(hh2 + hr2, 1e-15f));
        float stt = dev_artanh(ncat) / ncat;
        // pass 2: re-gather and emit vectors
        for (int d = 0; d < D_SZ; ++d) {
            float hv = E[eh0 + d]; if (layer == 1) hv += E[eh1 + d];
            float pv = R[er0 + d]; if (layer == 1) pv *= R[er1 + d];
            float tv = E[et + d];
            float hhv = scale_hh * hv;
            float htv = cA * hv + cB * tv;
            float hrv = cAr * hv + cBr * pv;
            sHH[m * 65 + d] = hhv;
            sHT[m * 65 + d] = htv;
            stang[m * XSTR + d]      = (_Float16)(stt * hhv);
            stang[m * XSTR + 64 + d] = (_Float16)(stt * hrv);
        }
    }
    __syncthreads();

    // a = tanh(relu(tang @ w1 + b1) @ w2 + b2), via WMMA
    int wv = t >> 5, lane = t & 31, row0 = wv * 16;
    float dot[8];
#pragma unroll
    for (int r = 0; r < 8; ++r) dot[r] = 0.f;
#pragma unroll
    for (int nt = 0; nt < 4; ++nt) {
        v8f acc = {};
#pragma unroll
        for (int kc = 0; kc < 4; ++kc) {
            v16h a  = load_a_frag(stang, row0, kc * 32, XSTR);
            v16h bb = load_b_frag_nk(sw1t, nt * 16, kc * 32, XSTR);
            acc = __builtin_amdgcn_wmma_f32_16x16x32_f16(false, a, false, bb,
                                                         (short)0, acc, false, false);
        }
        int col = nt * 16 + (lane & 15);
        float bias = sb1[col], wc = sw2[col];
#pragma unroll
        for (int r = 0; r < 8; ++r) dot[r] += fmaxf(acc[r] + bias, 0.f) * wc;
    }
    float b2v = b2p[0];
#pragma unroll
    for (int r = 0; r < 8; ++r) {
        float s = dot[r];
        s += __shfl_xor(s, 1, 32); s += __shfl_xor(s, 2, 32);
        s += __shfl_xor(s, 4, 32); s += __shfl_xor(s, 8, 32);
        if ((lane & 15) == 0)
            satt[row0 + r + ((lane & 16) ? 8 : 0)] = tanhf(s + b2v);
    }
    __syncthreads();

    if (t < 64) {
        float mx = -1e30f;
        for (int m = 0; m < M_SZ; ++m) mx = fmaxf(mx, satt[m]);
        float sum = 0.f;
        for (int m = 0; m < M_SZ; ++m) sum += expf(satt[m] - mx);
        swgt[t] = expf(satt[t] - mx) / sum;
    }
    __syncthreads();

    if (t < 64) {  // mean_pt[d] = sum_m w[m]*hh[m][d]
        float acc = 0.f;
        for (int m = 0; m < M_SZ; ++m) acc += swgt[m] * sHH[m * 65 + t];
        smean[t] = acc;
    }
    __syncthreads();

    if (t < 64) {  // e_m = logmap(w*ht_m, mean_pt); contribution w*e_m (stang storage reused)
        int m = t;
        float x2 = 0.f;
        for (int d = 0; d < D_SZ; ++d) x2 += smean[d] * smean[d];
        float pdot = 0.f;
        for (int d = 0; d < D_SZ; ++d) pdot += smean[d] * sHT[m * 65 + d];
        float w = swgt[m];
        float y2 = w * w * ht2_reg;
        float xy = -w * pdot;
        float den = fmaxf(1.f + 2.f * xy + x2 * y2, 1e-15f);
        float cn1 = (1.f + 2.f * xy + y2) / den;   // multiplies (-mean_pt)
        float cn2 = (1.f - x2) / den;              // multiplies (w*ht)
        float n2 = 0.f;
        for (int d = 0; d < D_SZ; ++d) {
            float mm = -cn1 * smean[d] + cn2 * w * sHT[m * 65 + d];
            n2 += mm * mm;
        }
        float n = sqrtf(fmaxf(n2, 1e-15f));
        float lamp = 2.f / fmaxf(1.f - x2, 1e-15f);
        float ce = (2.f / lamp) * dev_artanh(n) / n * w;  // includes outer w[m]
        for (int d = 0; d < D_SZ; ++d) {
            float mm = -cn1 * smean[d] + cn2 * w * sHT[m * 65 + d];
            sS[m * 65 + d] = ce * mm;
        }
    }
    __syncthreads();

    if (t < 64) {
        float acc = 0.f;
        for (int m = 0; m < M_SZ; ++m) acc += sS[m * 65 + t];
        out[(size_t)b * D_SZ + t] = acc;
    }
}

// ---------------------------------------------------------------------------
// mlcl support kernels
// ---------------------------------------------------------------------------
__global__ void k_l2rows(const float* __restrict__ in, float* __restrict__ outp) {
    int r = blockIdx.x * blockDim.x + threadIdx.x;  // ROWS threads
    if (r >= ROWS) return;
    const float* p = in + (size_t)r * D_SZ;
    float s = 0.f;
    for (int d = 0; d < D_SZ; ++d) s += p[d] * p[d];
    float inv = 1.f / fmaxf(sqrtf(s), 1e-12f);
    float* q = outp + (size_t)r * D_SZ;
    for (int d = 0; d < D_SZ; ++d) q[d] = p[d] * inv;
}

__global__ void k_pos(const float* __restrict__ na, const float* __restrict__ nb,
                      float* __restrict__ pos) {
    int r = blockIdx.x * blockDim.x + threadIdx.x;
    if (r >= ROWS) return;
    float s = 0.f;
    for (int d = 0; d < D_SZ; ++d) s += na[(size_t)r * D_SZ + d] * nb[(size_t)r * D_SZ + d];
    pos[r] = s;
}

__global__ void k_zero(float* __restrict__ p, int n) {
    int i = blockIdx.x * blockDim.x + threadIdx.x;
    if (i < n) p[i] = 0.f;
}

// G = NA @ NB^T (6144x6144), rowsum[r] = sum_m exp(G[r,m]/T),
// colsum[m] = sum_r exp(G[r,m]/T).  96 blocks x 128 thr, 64 rows per block.
__global__ void k_expsum(const float* __restrict__ NA, const float* __restrict__ NB,
                         float* __restrict__ rowsum, float* __restrict__ colsum) {
    __shared__ __align__(16) _Float16 sna[64 * NSTR];  // [row][k]
    __shared__ __align__(16) _Float16 snb[64 * NSTR];  // [col][k]
    __shared__ float srow[64];
    __shared__ float scol[64];

    int t = threadIdx.x;
    int r0g = blockIdx.x * 64;
    for (int i = t; i < 64 * 64; i += 128) {
        int r = i >> 6, k = i & 63;
        sna[r * NSTR + k] = (_Float16)NA[(size_t)(r0g + r) * D_SZ + k];
    }
    if (t < 64) srow[t] = 0.f;
    __syncthreads();

    int wv = t >> 5, lane = t & 31, row0 = wv * 16;
    for (int cb = 0; cb < 96; ++cb) {
        for (int i = t; i < 64 * 64; i += 128) {
            int n = i >> 6, k = i & 63;
            snb[n * NSTR + k] = (_Float16)NB[(size_t)(cb * 64 + n) * D_SZ + k];
        }
        if (t < 64) scol[t] = 0.f;
        __syncthreads();
#pragma unroll
        for (int nt = 0; nt < 4; ++nt) {
            v8f acc = {};
#pragma unroll
            for (int kc = 0; kc < 2; ++kc) {
                v16h a  = load_a_frag(sna, row0, kc * 32, NSTR);
                v16h bb = load_b_frag_nk(snb, nt * 16, kc * 32, NSTR);
                acc = __builtin_amdgcn_wmma_f32_16x16x32_f16(false, a, false, bb,
                                                             (short)0, acc, false, false);
            }
            float ev[8], ca = 0.f;
#pragma unroll
            for (int r = 0; r < 8; ++r) { ev[r] = expf(acc[r] * INV_T); ca += ev[r]; }
            ca += __shfl_xor(ca, 16, 32);                 // combine 16-row halves per col
            if (lane < 16) atomicAdd(&scol[nt * 16 + lane], ca);
#pragma unroll
            for (int r = 0; r < 8; ++r) {
                float s = ev[r];
                s += __shfl_xor(s, 1, 32); s += __shfl_xor(s, 2, 32);
                s += __shfl_xor(s, 4, 32); s += __shfl_xor(s, 8, 32);
                if ((lane & 15) == 0)
                    srow[row0 + r + ((lane & 16) ? 8 : 0)] += s;  // unique row owner
            }
        }
        __syncthreads();
        if (t < 64) atomicAdd(&colsum[cb * 64 + t], scol[t]);
        __syncthreads();
    }
    if (t < 64) rowsum[r0g + t] = srow[t];
}

// loss = KGE_W * sum_r [log rs + log cs - 2*pos/T] over both pairs
__global__ void k_loss(const float* __restrict__ rs1, const float* __restrict__ cs1,
                       const float* __restrict__ pos1,
                       const float* __restrict__ rs2, const float* __restrict__ cs2,
                       const float* __restrict__ pos2, float* __restrict__ out) {
    __shared__ float red[256];
    int t = threadIdx.x;
    float acc = 0.f;
    for (int r = t; r < ROWS; r += 256) {
        acc += logf(rs1[r]) + logf(cs1[r]) - 2.f * INV_T * pos1[r];
        acc += logf(rs2[r]) + logf(cs2[r]) - 2.f * INV_T * pos2[r];
    }
    red[t] = acc;
    __syncthreads();
    for (int s = 128; s > 0; s >>= 1) { if (t < s) red[t] += red[t + s]; __syncthreads(); }
    if (t == 0) out[0] = 1e-6f * red[0];
}

// scores[b] = sigmoid(sum_l ucf.icf + ukg.ikg)
__global__ void k_scores(const float* __restrict__ ucf, const float* __restrict__ ukg,
                         const float* __restrict__ ikg, const float* __restrict__ icf,
                         float* __restrict__ out) {
    int b = blockIdx.x * blockDim.x + threadIdx.x;
    if (b >= B_SZ) return;
    float s = 0.f;
    for (int l = 0; l < 3; ++l) {
        size_t off = ((size_t)l * B_SZ + b) * D_SZ;
        for (int d = 0; d < D_SZ; ++d)
            s += ucf[off + d] * icf[off + d] + ukg[off + d] * ikg[off + d];
    }
    out[b] = 1.f / (1.f + expf(-s));
}

// ---------------------------------------------------------------------------
extern "C" void kernel_launch(void* const* d_in, const int* in_sizes, int n_in,
                              void* d_out, int out_size, void* d_ws, size_t ws_size,
                              hipStream_t stream) {
    (void)in_sizes; (void)n_in; (void)out_size; (void)ws_size;
    const int* items = (const int*)d_in[0];
    const int* ucf_h = (const int*)d_in[1];
    const int* ucf_r = (const int*)d_in[2];
    const int* ucf_t = (const int*)d_in[3];
    const int* ikg_h = (const int*)d_in[4];
    const int* ikg_r = (const int*)d_in[5];
    const int* ikg_t = (const int*)d_in[6];
    const int* ukg_h = (const int*)d_in[7];
    const int* ukg_r = (const int*)d_in[8];
    const int* ukg_t = (const int*)d_in[9];
    const int* icf_h = (const int*)d_in[10];
    const int* icf_r = (const int*)d_in[11];
    const int* icf_t = (const int*)d_in[12];
    const float* E      = (const float*)d_in[13];
    const float* R      = (const float*)d_in[14];
    const float* att_w1 = (const float*)d_in[15];
    const float* att_w2 = (const float*)d_in[16];
    const float* a2w1   = (const float*)d_in[17];
    const float* a2b1   = (const float*)d_in[18];
    const float* a2w2   = (const float*)d_in[19];
    const float* a2b2   = (const float*)d_in[20];
    const float* curv   = (const float*)d_in[21];
    float* out = (float*)d_out;

    float* ws = (float*)d_ws;
    const size_t SS = (size_t)3 * B_SZ * D_SZ;    // one stack: 393216 floats
    float* ucf = ws;            float* ukg = ws + SS;
    float* ikg = ws + 2 * SS;   float* icf = ws + 3 * SS;
    float* nucf = ws + 4 * SS;  float* nukg = ws + 5 * SS;
    float* nikg = ws + 6 * SS;  float* nicf = ws + 7 * SS;
    float* rs1 = ws + 8 * SS;   float* cs1 = rs1 + ROWS;  float* pos1 = cs1 + ROWS;
    float* rs2 = pos1 + ROWS;   float* cs2 = rs2 + ROWS;  float* pos2 = cs2 + ROWS;

    dim3 gB(B_SZ);
    k_mean_init<<<gB, 64, 0, stream>>>(E, ucf_h, ucf);
    k_item_init<<<gB, 64, 0, stream>>>(E, items, ikg);
    k_mean_init<<<gB, 64, 0, stream>>>(E, ukg_h, ukg);
    k_mean_init<<<gB, 64, 0, stream>>>(E, icf_h, icf);

    const size_t LBD = (size_t)B_SZ * D_SZ;
    for (int layer = 0; layer < 2; ++layer) {
        k_ed_agg<<<gB, 128, 0, stream>>>(E, R, ucf_h, ucf_r, ucf_t, att_w1, att_w2,
                                         ucf + (size_t)(layer + 1) * LBD, layer);
        k_ed_agg<<<gB, 128, 0, stream>>>(E, R, ikg_h, ikg_r, ikg_t, att_w1, att_w2,
                                         ikg + (size_t)(layer + 1) * LBD, layer);
        k_hyper_agg<<<gB, 128, 0, stream>>>(E, R, ukg_h, ukg_r, ukg_t, a2w1, a2b1, a2w2,
                                            a2b2, curv, ukg + (size_t)(layer + 1) * LBD, layer);
        k_hyper_agg<<<gB, 128, 0, stream>>>(E, R, icf_h, icf_r, icf_t, a2w1, a2b1, a2w2,
                                            a2b2, curv, icf + (size_t)(layer + 1) * LBD, layer);
    }

    k_l2rows<<<48, 128, 0, stream>>>(ucf, nucf);
    k_l2rows<<<48, 128, 0, stream>>>(ukg, nukg);
    k_l2rows<<<48, 128, 0, stream>>>(ikg, nikg);
    k_l2rows<<<48, 128, 0, stream>>>(icf, nicf);

    k_zero<<<(6 * ROWS + 255) / 256, 256, 0, stream>>>(rs1, 6 * ROWS);
    k_pos<<<48, 128, 0, stream>>>(nucf, nukg, pos1);
    k_pos<<<48, 128, 0, stream>>>(nikg, nicf, pos2);
    k_expsum<<<96, 128, 0, stream>>>(nucf, nukg, rs1, cs1);
    k_expsum<<<96, 128, 0, stream>>>(nikg, nicf, rs2, cs2);

    k_loss<<<1, 256, 0, stream>>>(rs1, cs1, pos1, rs2, cs2, pos2, out + B_SZ);
    k_scores<<<16, 128, 0, stream>>>(ucf, ukg, ikg, icf, out);
}
